// Model_44255343018782
// MI455X (gfx1250) — matrix operand order repaired and checked by reading, not compiled, not run on hardware
//
#include <hip/hip_runtime.h>

#define D_IN   128
#define D_MAIN 256
#define D_BLK  512
#define NCAND  100000
#define NCANDP 100096          // 782 * 128
#define NTC    (NCANDP / 16)   // 6256 candidate N-tiles
#define NSHARD 8
#define SHT    (NTC / NSHARD)  // 782 tiles per shard
#define NBATCH 1024
#define CTX    96

typedef __attribute__((ext_vector_type(16))) _Float16 v16h;
typedef __attribute__((ext_vector_type(8)))  float    v8f;

static __device__ __forceinline__ v8f wmma_f16(v16h a, v16h b, v8f c) {
  // D = A(16x32,f16) * B(32x16,f16) + C(16x16,f32)
  return __builtin_amdgcn_wmma_f32_16x16x32_f16(false, a, false, b, (short)0, c, false, false);
}

// ISA 7.12.2 16-bit A layout: lane = M (mod 16); halves 0..7 -> K pairs
// {0..7 | 16..23}; hi lanes add +8.
static __device__ __forceinline__ int a_koff(int lane, int i) {
  const int pair = i >> 1, p = i & 1;
  const int kk = (pair < 4) ? (2 * pair + p) : (16 + 2 * (pair - 4) + p);
  return kk + ((lane >= 16) ? 8 : 0);
}
// B layout: lane = N (mod 16); halves 0..7 -> K {0..7 | 16..23}; hi lanes add +8 (same as A).
static __device__ __forceinline__ int b_koff(int lane, int i) {
  return (i < 8 ? i : i + 8) + ((lane >= 16) ? 8 : 0);
}

// A fragment (16x32) gathered from a row-major f16 LDS tile.
static __device__ __forceinline__ v16h load_a(const _Float16* base, int ld, int kbase, int lane) {
  const int m = lane & 15;
  v16h a;
#pragma unroll
  for (int i = 0; i < 16; ++i) a[i] = base[m * ld + kbase + a_koff(lane, i)];
  return a;
}

// Pre-swizzled fragment fetch: one 32-byte per-lane load (2x global_load_b128).
static __device__ __forceinline__ v16h load_frag(const _Float16* F, size_t tile, int lane) {
  return ((const v16h*)F)[tile * 32 + lane];
}

// ---------------- weight pre-swizzle: W[K][N] f32 -> B-fragment f16 -------------
__global__ __launch_bounds__(256) void prep_w_kernel(const float* __restrict__ W,
                                                     int K, int N, _Float16* __restrict__ F) {
  const int lane = threadIdx.x & 31, wid = threadIdx.x >> 5;
  const int NT = N / 16;
  const int tiles = (K / 32) * NT;
  const int tile = blockIdx.x * 8 + wid;
  if (tile >= tiles) return;
  const int kt = tile / NT, nt = tile % NT;
  const int n  = nt * 16 + (lane & 15);
  const int k0 = kt * 32;
  _Float16* dst = F + (size_t)tile * 512 + lane * 16;
#pragma unroll
  for (int i = 0; i < 16; ++i) dst[i] = (_Float16)W[(size_t)(k0 + b_koff(lane, i)) * N + n];
}

// ---------------- encoder: x -> (x_enc, k) for a 16-row tile --------------------
__global__ __launch_bounds__(256) void encode_kernel(
    const float* __restrict__ xin, int nrows,
    const _Float16* __restrict__ fLin, const float* __restrict__ lin_b,
    const _Float16* __restrict__ fW1, const float* __restrict__ b1,
    const _Float16* __restrict__ fW2, const float* __restrict__ b2,
    const float* __restrict__ mix_g, const float* __restrict__ mix_b,
    const _Float16* __restrict__ fKw, const float* __restrict__ K_b,
    float* __restrict__ out_x, _Float16* __restrict__ out_k16,
    float* __restrict__ out_k32, float* __restrict__ out_norm,
    _Float16* __restrict__ out_ckfrag,   // candidate k in B-fragment layout (or null)
    _Float16* __restrict__ out_kafrag)   // batch k in A-fragment layout (or null)
{
  const int LDA = D_MAIN + 8;
  const int LDU = D_BLK + 8;
  __shared__ _Float16 aS[16 * (D_MAIN + 8)];
  __shared__ _Float16 uS[16 * (D_BLK + 8)];
  __shared__ float hS[16 * D_MAIN];
  __shared__ float ps[256], ps2[256];
  __shared__ float redm[16], redv[16];
  const int tid = threadIdx.x, lane = tid & 31, wid = tid >> 5;
  const int row0 = blockIdx.x * 16;
  const int col = lane & 15;
  const int mh = (lane >= 16) ? 8 : 0;

  // stage input rows (f32 -> f16)
  for (int i = tid; i < 16 * D_IN; i += 256) {
    const int r = i >> 7, c = i & (D_IN - 1);
    const float v = (row0 + r < nrows) ? xin[(size_t)(row0 + r) * D_IN + c] : 0.f;
    aS[r * LDA + c] = (_Float16)v;
  }
  __syncthreads();

  // h1 = x @ lin_w + lin_b
#pragma unroll
  for (int t = 0; t < 2; ++t) {
    const int nt = wid * 2 + t, nb = nt * 16;
    v8f acc = {};
#pragma unroll
    for (int ks = 0; ks < D_IN / 32; ++ks)
      acc = wmma_f16(load_a(aS, LDA, ks * 32, lane), load_frag(fLin, ks * 16 + nt, lane), acc);
    const float bias = lin_b[nb + col];
#pragma unroll
    for (int r = 0; r < 8; ++r) hS[(r + mh) * D_MAIN + nb + col] = acc[r] + bias;
  }
  __syncthreads();

  for (int i = tid; i < 16 * D_MAIN; i += 256) {
    const int r = i >> 8, c = i & (D_MAIN - 1);
    aS[r * LDA + c] = (_Float16)hS[i];
  }
  __syncthreads();

  // u = relu(h1 @ w1 + b1)
#pragma unroll
  for (int t = 0; t < 4; ++t) {
    const int nt = wid * 4 + t, nb = nt * 16;
    v8f acc = {};
#pragma unroll
    for (int ks = 0; ks < D_MAIN / 32; ++ks)
      acc = wmma_f16(load_a(aS, LDA, ks * 32, lane), load_frag(fW1, ks * 32 + nt, lane), acc);
    const float bias = b1[nb + col];
#pragma unroll
    for (int r = 0; r < 8; ++r) {
      const float v = acc[r] + bias;
      uS[(r + mh) * LDU + nb + col] = (_Float16)(v > 0.f ? v : 0.f);
    }
  }
  __syncthreads();

  // h2 = u @ w2 + b2 + h1 (residual; same-lane read->write of hS)
#pragma unroll
  for (int t = 0; t < 2; ++t) {
    const int nt = wid * 2 + t, nb = nt * 16;
    v8f acc = {};
#pragma unroll
    for (int ks = 0; ks < D_BLK / 32; ++ks)
      acc = wmma_f16(load_a(uS, LDU, ks * 32, lane), load_frag(fW2, ks * 16 + nt, lane), acc);
    const float bias = b2[nb + col];
#pragma unroll
    for (int r = 0; r < 8; ++r) {
      const int m = r + mh, cidx = nb + col;
      const float v = acc[r] + bias + hS[m * D_MAIN + cidx];
      hS[m * D_MAIN + cidx] = v;
      if (out_x) out_x[(size_t)(row0 + m) * D_MAIN + cidx] = v;
    }
  }
  __syncthreads();

  // LayerNorm(mix_g, mix_b)
  {
    const int r = tid >> 4, s = tid & 15;
    float ls = 0.f, lq = 0.f;
#pragma unroll
    for (int q = 0; q < 16; ++q) { const float v = hS[r * D_MAIN + s * 16 + q]; ls += v; lq += v * v; }
    ps[tid] = ls; ps2[tid] = lq;
    __syncthreads();
    if (s == 0) {
      float a = 0.f, bb = 0.f;
      for (int q = 0; q < 16; ++q) { a += ps[r * 16 + q]; bb += ps2[r * 16 + q]; }
      const float mean = a * (1.f / D_MAIN);
      const float var  = bb * (1.f / D_MAIN) - mean * mean;
      redm[r] = mean; redv[r] = rsqrtf(var + 1e-5f);
    }
  }
  __syncthreads();
  for (int i = tid; i < 16 * D_MAIN; i += 256) {
    const int r = i >> 8, c = i & (D_MAIN - 1);
    aS[r * LDA + c] = (_Float16)((hS[i] - redm[r]) * redv[r] * mix_g[c] + mix_b[c]);
  }
  __syncthreads();

  // k = LN(h2) @ K_w + K_b
#pragma unroll
  for (int t = 0; t < 2; ++t) {
    const int nt = wid * 2 + t, nb = nt * 16;
    v8f acc = {};
#pragma unroll
    for (int ks = 0; ks < D_MAIN / 32; ++ks)
      acc = wmma_f16(load_a(aS, LDA, ks * 32, lane), load_frag(fKw, ks * 16 + nt, lane), acc);
    const float bias = K_b[nb + col];
#pragma unroll
    for (int r = 0; r < 8; ++r) {
      const int m = r + mh, cidx = nb + col;
      const float v = acc[r] + bias;
      hS[m * D_MAIN + cidx] = v;
      const size_t g = (size_t)(row0 + m) * D_MAIN + cidx;
      out_k16[g] = (_Float16)v;
      if (out_k32) out_k32[g] = v;
    }
  }
  __syncthreads();

  // emit k in dist-GEMM fragment layouts straight from LDS (coalesced 32B/lane)
  if (out_ckfrag) {   // B-fragment: this block is candidate N-tile blockIdx.x; wave = kt
    _Float16* dst = out_ckfrag + ((size_t)wid * NTC + blockIdx.x) * 512 + lane * 16;
#pragma unroll
    for (int i = 0; i < 16; ++i)
      dst[i] = (_Float16)hS[(lane & 15) * D_MAIN + wid * 32 + b_koff(lane, i)];
  }
  if (out_kafrag) {   // A-fragment: this block is batch M-tile blockIdx.x; wave = kt
    _Float16* dst = out_kafrag + ((size_t)blockIdx.x * 8 + wid) * 512 + lane * 16;
#pragma unroll
    for (int i = 0; i < 16; ++i)
      dst[i] = (_Float16)hS[(lane & 15) * D_MAIN + wid * 32 + a_koff(lane, i)];
  }

  // squared norms ||k||^2 per row
  {
    const int r = tid >> 4, s = tid & 15;
    float lq = 0.f;
#pragma unroll
    for (int q = 0; q < 16; ++q) { const float v = hS[r * D_MAIN + s * 16 + q]; lq += v * v; }
    ps[tid] = lq;
    __syncthreads();
    if (s == 0) {
      float a = 0.f;
      for (int q = 0; q < 16; ++q) a += ps[r * 16 + q];
      out_norm[row0 + r] = a;
    }
  }
}

// 16 parallel bitonic sorts of 256 (key,idx) pairs, one per batch row of the tile.
static __device__ void rowsort256(unsigned* keyB, unsigned* idxB, const int* cntS) {
  const int tid = threadIdx.x;
  for (int e = tid; e < 16 * 256; e += 256) {
    const int row = e >> 8, t = e & 255;
    if (t >= cntS[row]) { keyB[e] = 0xFFFFFFFFu; idxB[e] = 0xFFFFFFFFu; }
  }
  __syncthreads();
  for (int k = 2; k <= 256; k <<= 1) {
    for (int j = k >> 1; j > 0; j >>= 1) {
      for (int e = tid; e < 16 * 256; e += 256) {
        const int row = e >> 8, t = e & 255;
        const int l = t ^ j;
        if (l > t) {
          const int bi = row << 8;
          const unsigned ka = keyB[bi + t], kb = keyB[bi + l];
          const unsigned ia = idxB[bi + t], ib = idxB[bi + l];
          const bool agtb = (ka > kb) || (ka == kb && ia > ib);
          if (agtb == ((t & k) == 0)) {
            keyB[bi + t] = kb; keyB[bi + l] = ka;
            idxB[bi + t] = ib; idxB[bi + l] = ia;
          }
        }
      }
      __syncthreads();
    }
  }
}

// ---------------- fused dist GEMM + per-row top-96 (per candidate shard) --------
// dist[b][j] = ||k_b||^2 - 2 k_b.ck_j + ||ck_j||^2 computed in WMMA accumulators,
// filtered straight from registers into per-row LDS buffers. Distances never
// touch global memory: the sortable key is exactly invertible.
__global__ __launch_bounds__(256) void distsel_kernel(
    const _Float16* __restrict__ kbA, const _Float16* __restrict__ ckF,
    const float* __restrict__ knorm, const float* __restrict__ cnorm,
    unsigned* __restrict__ pkey, unsigned* __restrict__ pidx)
{
  __shared__ unsigned keyB[16 * 256];
  __shared__ unsigned idxB[16 * 256];
  __shared__ int cntS[16];
  __shared__ unsigned thrS[16];
  __shared__ int flagS;
  const int tid = threadIdx.x, lane = tid & 31, wid = tid >> 5;
  const int bt = blockIdx.x;            // batch row-tile
  const int sh = blockIdx.y;            // candidate shard
  const int b0 = bt * 16;
  const int mh = (lane >= 16) ? 8 : 0;

  if (tid < 16) { cntS[tid] = 0; thrS[tid] = 0xFFFFFFFFu; }
  if (tid == 0) flagS = 0;

  // A fragments + row norms resident in registers for the whole shard
  v16h afr[8];
#pragma unroll
  for (int ks = 0; ks < 8; ++ks) afr[ks] = load_frag(kbA, (size_t)bt * 8 + ks, lane);
  float kn[8];
#pragma unroll
  for (int r = 0; r < 8; ++r) kn[r] = knorm[b0 + r + mh];
  __syncthreads();

  const int t0 = sh * SHT;
  for (int it = 0; it < (SHT + 7) / 8; ++it) {
    const int jt = t0 + it * 8 + wid;
    if (jt < t0 + SHT) {
      if (it * 8 + wid + 8 < SHT)       // gfx1250 global_prefetch_b8, one tile ahead
        __builtin_prefetch((const char*)ckF + (size_t)(jt + 8) * 1024, 0, 1);
      v8f acc = {};
#pragma unroll
      for (int ks = 0; ks < 8; ++ks)
        acc = wmma_f16(afr[ks], load_frag(ckF, (size_t)ks * NTC + jt, lane), acc);
      const int j = jt * 16 + (lane & 15);
      if (j < NCAND) {
        const float cn = cnorm[j];
#pragma unroll
        for (int r = 0; r < 8; ++r) {
          const int m = r + mh;
          const float d = kn[r] - 2.f * acc[r] + cn;
          unsigned u = __float_as_uint(d);
          u = (u & 0x80000000u) ? ~u : (u | 0x80000000u);   // sortable ascending
          if (u < thrS[m]) {
            const int p = atomicAdd(&cntS[m], 1);
            if (p < 256) { keyB[m * 256 + p] = u; idxB[m * 256 + p] = (unsigned)j; }
            if (p >= 128) flagS = 1;    // benign set-only race
          }
        }
      }
    }
    __syncthreads();
    if (flagS) {                        // uniform after barrier
      rowsort256(keyB, idxB, cntS);     // ends with a barrier
      if (tid < 16) { thrS[tid] = keyB[tid * 256 + CTX - 1]; cntS[tid] = CTX; }
      if (tid == 0) flagS = 0;
      __syncthreads();
    }
  }
  rowsort256(keyB, idxB, cntS);
  for (int e = tid; e < 16 * CTX; e += 256) {
    const int row = e / CTX, q = e % CTX;
    const size_t o = (((size_t)bt * NSHARD + sh) * 16 + row) * CTX + q;
    pkey[o] = keyB[row * 256 + q];
    pidx[o] = idxB[row * 256 + q];
  }
}

// ---------------- merge 8 sorted shard-partials -> global top-96 per row --------
__global__ __launch_bounds__(256) void merge_kernel(
    const unsigned* __restrict__ pkey, const unsigned* __restrict__ pidx,
    int* __restrict__ ctx_idx, float* __restrict__ ctx_dist)
{
  __shared__ unsigned keyS[1024], idxS[1024];
  const int tid = threadIdx.x;
  const size_t b = blockIdx.x;
  const size_t bt = b >> 4, r = b & 15;
  for (int e = tid; e < 1024; e += 256) {
    if (e < NSHARD * CTX) {
      const int s = e / CTX, q = e % CTX;
      const size_t o = ((bt * NSHARD + s) * 16 + r) * CTX + q;
      keyS[e] = pkey[o]; idxS[e] = pidx[o];
    } else { keyS[e] = 0xFFFFFFFFu; idxS[e] = 0xFFFFFFFFu; }
  }
  __syncthreads();
  for (int k = 2; k <= 1024; k <<= 1) {
    for (int j = k >> 1; j > 0; j >>= 1) {
      for (int t = tid; t < 1024; t += 256) {
        const int l = t ^ j;
        if (l > t) {
          const unsigned ka = keyS[t], kb = keyS[l];
          const unsigned ia = idxS[t], ib = idxS[l];
          const bool agtb = (ka > kb) || (ka == kb && ia > ib);
          if (agtb == ((t & k) == 0)) { keyS[t] = kb; keyS[l] = ka; idxS[t] = ib; idxS[l] = ia; }
        }
      }
      __syncthreads();
    }
  }
  if (tid < CTX) {
    const unsigned u = keyS[tid];
    const float f = (u & 0x80000000u) ? __uint_as_float(u ^ 0x80000000u) : __uint_as_float(~u);
    ctx_idx[b * CTX + tid]  = (int)idxS[tid];   // ascending dist == top_k(-dist) order
    ctx_dist[b * CTX + tid] = f;
  }
}

// ---------------- per-row context: softmax + T-MLP + aggregation ----------------
__global__ __launch_bounds__(256) void context_kernel(
    const float* __restrict__ kb32, const _Float16* __restrict__ ck16,
    const float* __restrict__ ctx_dist, const int* __restrict__ ctx_idx,
    const float* __restrict__ cand_y,
    const float* __restrict__ lab_w, const float* __restrict__ lab_b,
    const _Float16* __restrict__ fT1, const float* __restrict__ T_b1,
    const _Float16* __restrict__ fT2,
    const float* __restrict__ xb, float* __restrict__ xa)
{
  const int LDA = D_MAIN + 4;   // 260
  const int LDH = 66;
  __shared__ _Float16 aS[CTX * (D_MAIN + 4)];   // diff matrix, f16
  __shared__ _Float16 hS[CTX * 66];             // 64-col H chunk, f16
  __shared__ int   idxS[CTX];
  __shared__ float probS[CTX];
  __shared__ float simS[CTX];
  __shared__ float pyS;
  const int tid = threadIdx.x, lane = tid & 31, wid = tid >> 5;
  const size_t b = blockIdx.x;
  const int col = lane & 15, mh = (lane >= 16) ? 8 : 0;

  if (tid < CTX) {
    idxS[tid] = ctx_idx[b * CTX + tid];
    simS[tid] = -ctx_dist[b * CTX + tid];
  }
  __syncthreads();
  if (tid == 0) {
    float mx = -3.4e38f;
    for (int c = 0; c < CTX; ++c) mx = fmaxf(mx, simS[c]);
    float s = 0.f;
    for (int c = 0; c < CTX; ++c) { const float e = __expf(simS[c] - mx); probS[c] = e; s += e; }
    const float inv = 1.f / s;
    float py = 0.f;
    for (int c = 0; c < CTX; ++c) { probS[c] *= inv; py += probS[c] * cand_y[idxS[c]]; }
    pyS = py;
  }
  // diff = k[b] - ck[idx[c]]
  for (int i = tid; i < CTX * D_MAIN; i += 256) {
    const int c = i >> 8, d = i & (D_MAIN - 1);
    const float v = kb32[b * D_MAIN + d] - (float)ck16[(size_t)idxS[c] * D_MAIN + d];
    aS[c * LDA + d] = (_Float16)v;
  }
  __syncthreads();

  v8f vacc[6][2];
  const v8f vz = {};
#pragma unroll
  for (int mt = 0; mt < 6; ++mt)
#pragma unroll
    for (int t = 0; t < 2; ++t) vacc[mt][t] = vz;

  for (int kc = 0; kc < D_BLK / 64; ++kc) {
    // H chunk = relu(diff @ T_w1[:, kc*64 : kc*64+64] + b)  -> hS (f16)
    for (int job = wid; job < 24; job += 8) {
      const int mt = job >> 2, ntc = job & 3;
      const int ntG = kc * 4 + ntc;               // N-tile within D_BLK
      const int nbG = ntG * 16;
      v8f acc = {};
#pragma unroll
      for (int ks = 0; ks < 8; ++ks)
        acc = wmma_f16(load_a(aS + mt * 16 * LDA, LDA, ks * 32, lane),
                       load_frag(fT1, (size_t)ks * 32 + ntG, lane), acc);
      const float bias = T_b1[nbG + col];
#pragma unroll
      for (int r = 0; r < 8; ++r) {
        const float v = acc[r] + bias;
        hS[(mt * 16 + r + mh) * LDH + ntc * 16 + col] = (_Float16)(v > 0.f ? v : 0.f);
      }
    }
    __syncthreads();
    // V += H_chunk @ T_w2[kc*64 : kc*64+64, :]
#pragma unroll
    for (int mt = 0; mt < 6; ++mt) {
#pragma unroll
      for (int t = 0; t < 2; ++t) {
        const int nt = wid * 2 + t;
#pragma unroll
        for (int ks = 0; ks < 2; ++ks)
          vacc[mt][t] = wmma_f16(load_a(hS + mt * 16 * LDH, LDH, ks * 32, lane),
                                 load_frag(fT2, (size_t)(kc * 2 + ks) * 16 + nt, lane), vacc[mt][t]);
      }
    }
    __syncthreads();
  }

  // context_x[d] = sum_c p_c * V[c][d] + py*lab_w[d] + lab_b[d]; xa = x + context_x
#pragma unroll
  for (int t = 0; t < 2; ++t) {
    const int nb = (wid * 2 + t) * 16;
    float part = 0.f;
#pragma unroll
    for (int mt = 0; mt < 6; ++mt)
#pragma unroll
      for (int r = 0; r < 8; ++r)
        part += probS[mt * 16 + r + mh] * vacc[mt][t][r];
    part += __shfl_xor(part, 16);     // combine the two M-halves of each column
    if (lane < 16) {
      const int c = nb + lane;
      xa[b * D_MAIN + c] = xb[b * D_MAIN + c] + part + pyS * lab_w[c] + lab_b[c];
    }
  }
}

// ---------------- final prenorm MLP + head --------------------------------------
__global__ __launch_bounds__(256) void final_kernel(
    const float* __restrict__ xa,
    const float* __restrict__ p_g, const float* __restrict__ p_b,
    const _Float16* __restrict__ fP1, const float* __restrict__ p_b1,
    const _Float16* __restrict__ fP2, const float* __restrict__ p_b2,
    const float* __restrict__ h_g, const float* __restrict__ h_b,
    const float* __restrict__ h_w, const float* __restrict__ h_b2,
    float* __restrict__ out)
{
  const int LDA = D_MAIN + 8, LDU = D_BLK + 8;
  __shared__ _Float16 aS[16 * (D_MAIN + 8)];
  __shared__ _Float16 uS[16 * (D_BLK + 8)];
  __shared__ float hS[16 * D_MAIN];
  __shared__ float ps[256], ps2[256];
  __shared__ float redm[16], redv[16];
  const int tid = threadIdx.x, lane = tid & 31, wid = tid >> 5;
  const int row0 = blockIdx.x * 16;
  const int col = lane & 15, mh = (lane >= 16) ? 8 : 0;

  for (int i = tid; i < 16 * D_MAIN; i += 256) hS[i] = xa[(size_t)row0 * D_MAIN + i];
  __syncthreads();

  // LN(p_g, p_b)
  {
    const int r = tid >> 4, s = tid & 15;
    float ls = 0.f, lq = 0.f;
#pragma unroll
    for (int q = 0; q < 16; ++q) { const float v = hS[r * D_MAIN + s * 16 + q]; ls += v; lq += v * v; }
    ps[tid] = ls; ps2[tid] = lq;
    __syncthreads();
    if (s == 0) {
      float a = 0.f, bb = 0.f;
      for (int q = 0; q < 16; ++q) { a += ps[r * 16 + q]; bb += ps2[r * 16 + q]; }
      const float mean = a * (1.f / D_MAIN);
      const float var  = bb * (1.f / D_MAIN) - mean * mean;
      redm[r] = mean; redv[r] = rsqrtf(var + 1e-5f);
    }
  }
  __syncthreads();
  for (int i = tid; i < 16 * D_MAIN; i += 256) {
    const int r = i >> 8, c = i & (D_MAIN - 1);
    aS[r * LDA + c] = (_Float16)((hS[i] - redm[r]) * redv[r] * p_g[c] + p_b[c]);
  }
  __syncthreads();

  // u = relu(LN @ p_w1 + p_b1)
#pragma unroll
  for (int t = 0; t < 4; ++t) {
    const int nt = wid * 4 + t, nb = nt * 16;
    v8f acc = {};
#pragma unroll
    for (int ks = 0; ks < 8; ++ks)
      acc = wmma_f16(load_a(aS, LDA, ks * 32, lane), load_frag(fP1, (size_t)ks * 32 + nt, lane), acc);
    const float bias = p_b1[nb + col];
#pragma unroll
    for (int r = 0; r < 8; ++r) {
      const float v = acc[r] + bias;
      uS[(r + mh) * LDU + nb + col] = (_Float16)(v > 0.f ? v : 0.f);
    }
  }
  __syncthreads();

  // y = u @ p_w2 + p_b2 + x
#pragma unroll
  for (int t = 0; t < 2; ++t) {
    const int nt = wid * 2 + t, nb = nt * 16;
    v8f acc = {};
#pragma unroll
    for (int ks = 0; ks < 16; ++ks)
      acc = wmma_f16(load_a(uS, LDU, ks * 32, lane), load_frag(fP2, (size_t)ks * 16 + nt, lane), acc);
    const float bias = p_b2[nb + col];
#pragma unroll
    for (int r = 0; r < 8; ++r) {
      const int m = r + mh, cidx = nb + col;
      hS[m * D_MAIN + cidx] = acc[r] + bias + hS[m * D_MAIN + cidx];
    }
  }
  __syncthreads();

  // head: relu(LN(y)) @ h_w + h_b2
  {
    const int r = tid >> 4, s = tid & 15;
    float ls = 0.f, lq = 0.f;
#pragma unroll
    for (int q = 0; q < 16; ++q) { const float v = hS[r * D_MAIN + s * 16 + q]; ls += v; lq += v * v; }
    ps[tid] = ls; ps2[tid] = lq;
    __syncthreads();
    if (s == 0) {
      float a = 0.f, bb = 0.f;
      for (int q = 0; q < 16; ++q) { a += ps[r * 16 + q]; bb += ps2[r * 16 + q]; }
      const float mean = a * (1.f / D_MAIN);
      const float var  = bb * (1.f / D_MAIN) - mean * mean;
      redm[r] = mean; redv[r] = rsqrtf(var + 1e-5f);
    }
    __syncthreads();
    float part = 0.f;
#pragma unroll
    for (int q = 0; q < 16; ++q) {
      const int c = s * 16 + q;
      float v = (hS[r * D_MAIN + c] - redm[r]) * redv[r] * h_g[c] + h_b[c];
      part += (v > 0.f ? v : 0.f) * h_w[c];
    }
    ps[tid] = part;
    __syncthreads();
    if (s == 0) {
      float a = 0.f;
      for (int q = 0; q < 16; ++q) a += ps[r * 16 + q];
      out[row0 + r] = a + h_b2[0];
    }
  }
}

// -------------------------------- launcher --------------------------------------
extern "C" void kernel_launch(void* const* d_in, const int* in_sizes, int n_in,
                              void* d_out, int out_size, void* d_ws, size_t ws_size,
                              hipStream_t stream) {
  (void)in_sizes; (void)n_in; (void)out_size; (void)ws_size;
  const float* x_num  = (const float*)d_in[0];
  const float* cand_x = (const float*)d_in[1];
  const float* cand_y = (const float*)d_in[2];
  // d_in[3] = context_size (fixed CTX = 96)
  const float* lin_w = (const float*)d_in[4];
  const float* lin_b = (const float*)d_in[5];
  const float* b0_w1 = (const float*)d_in[6];
  const float* b0_b1 = (const float*)d_in[7];
  const float* b0_w2 = (const float*)d_in[8];
  const float* b0_b2 = (const float*)d_in[9];
  const float* mix_g = (const float*)d_in[10];
  const float* mix_b = (const float*)d_in[11];
  const float* K_w   = (const float*)d_in[12];
  const float* K_b   = (const float*)d_in[13];
  const float* lab_w = (const float*)d_in[14];
  const float* lab_b = (const float*)d_in[15];
  const float* T_w1  = (const float*)d_in[16];
  const float* T_b1  = (const float*)d_in[17];
  const float* T_w2  = (const float*)d_in[18];
  const float* p_g   = (const float*)d_in[19];
  const float* p_b   = (const float*)d_in[20];
  const float* p_w1  = (const float*)d_in[21];
  const float* p_b1  = (const float*)d_in[22];
  const float* p_w2  = (const float*)d_in[23];
  const float* p_b2  = (const float*)d_in[24];
  const float* h_g   = (const float*)d_in[25];
  const float* h_b   = (const float*)d_in[26];
  const float* h_w   = (const float*)d_in[27];
  const float* h_b2  = (const float*)d_in[28];

  char* w = (char*)d_ws;
  size_t off = 0;
  auto take = [&](size_t bytes) { char* p = w + off; off = (off + bytes + 255) & ~(size_t)255; return p; };
  _Float16* ck16 = (_Float16*)take((size_t)NCANDP * D_MAIN * 2);   // row-major (context gather)
  _Float16* ckF  = (_Float16*)take((size_t)NCANDP * D_MAIN * 2);   // B-fragment layout (dist)
  _Float16* kbA  = (_Float16*)take((size_t)NBATCH * D_MAIN * 2);   // A-fragment layout (dist)
  _Float16* kb16 = (_Float16*)take((size_t)NBATCH * D_MAIN * 2);
  float* cnorm   = (float*)take((size_t)NCANDP * 4);
  float* kb32    = (float*)take((size_t)NBATCH * D_MAIN * 4);
  float* knorm   = (float*)take((size_t)NBATCH * 4);
  float* xb      = (float*)take((size_t)NBATCH * D_MAIN * 4);
  unsigned* pkey = (unsigned*)take((size_t)(NBATCH / 16) * NSHARD * 16 * CTX * 4);
  unsigned* pidx = (unsigned*)take((size_t)(NBATCH / 16) * NSHARD * 16 * CTX * 4);
  int*   ctxI    = (int*)take((size_t)NBATCH * CTX * 4);
  float* ctxD    = (float*)take((size_t)NBATCH * CTX * 4);
  float* xaB     = (float*)take((size_t)NBATCH * D_MAIN * 4);
  // pre-swizzled f16 weights (B-fragment layout, 1KB per 32x16 tile)
  _Float16* fLin = (_Float16*)take((size_t)(D_IN / 32) * (D_MAIN / 16) * 1024);
  _Float16* fW1  = (_Float16*)take((size_t)(D_MAIN / 32) * (D_BLK / 16) * 1024);
  _Float16* fW2  = (_Float16*)take((size_t)(D_BLK / 32) * (D_MAIN / 16) * 1024);
  _Float16* fKw  = (_Float16*)take((size_t)(D_MAIN / 32) * (D_MAIN / 16) * 1024);
  _Float16* fT1  = (_Float16*)take((size_t)(D_MAIN / 32) * (D_BLK / 16) * 1024);
  _Float16* fT2  = (_Float16*)take((size_t)(D_BLK / 32) * (D_MAIN / 16) * 1024);
  _Float16* fP1  = (_Float16*)take((size_t)(D_MAIN / 32) * (D_BLK / 16) * 1024);
  _Float16* fP2  = (_Float16*)take((size_t)(D_BLK / 32) * (D_MAIN / 16) * 1024);

  prep_w_kernel<<<8,  256, 0, stream>>>(lin_w, D_IN,   D_MAIN, fLin);
  prep_w_kernel<<<32, 256, 0, stream>>>(b0_w1, D_MAIN, D_BLK,  fW1);
  prep_w_kernel<<<32, 256, 0, stream>>>(b0_w2, D_BLK,  D_MAIN, fW2);
  prep_w_kernel<<<16, 256, 0, stream>>>(K_w,   D_MAIN, D_MAIN, fKw);
  prep_w_kernel<<<32, 256, 0, stream>>>(T_w1,  D_MAIN, D_BLK,  fT1);
  prep_w_kernel<<<32, 256, 0, stream>>>(T_w2,  D_BLK,  D_MAIN, fT2);
  prep_w_kernel<<<32, 256, 0, stream>>>(p_w1,  D_MAIN, D_BLK,  fP1);
  prep_w_kernel<<<32, 256, 0, stream>>>(p_w2,  D_BLK,  D_MAIN, fP2);

  encode_kernel<<<NCANDP / 16, 256, 0, stream>>>(cand_x, NCAND, fLin, lin_b, fW1, b0_b1,
                                                 fW2, b0_b2, mix_g, mix_b, fKw, K_b,
                                                 nullptr, ck16, nullptr, cnorm, ckF, nullptr);
  encode_kernel<<<NBATCH / 16, 256, 0, stream>>>(x_num, NBATCH, fLin, lin_b, fW1, b0_b1,
                                                 fW2, b0_b2, mix_g, mix_b, fKw, K_b,
                                                 xb, kb16, kb32, knorm, nullptr, kbA);
  distsel_kernel<<<dim3(NBATCH / 16, NSHARD), 256, 0, stream>>>(kbA, ckF, knorm, cnorm, pkey, pidx);
  merge_kernel<<<NBATCH, 256, 0, stream>>>(pkey, pidx, ctxI, ctxD);
  context_kernel<<<NBATCH, 256, 0, stream>>>(kb32, ck16, ctxD, ctxI, cand_y, lab_w, lab_b,
                                             fT1, T_b1, fT2, xb, xaB);
  final_kernel<<<NBATCH / 16, 256, 0, stream>>>(xaB, p_g, p_b, fP1, p_b1, fP2, p_b2,
                                                h_g, h_b, h_w, h_b2, (float*)d_out);
}
